// BrocaModule_11596411699834
// MI455X (gfx1250) — matrix operand order repaired and checked by reading, not compiled, not run on hardware
//
#include <hip/hip_runtime.h>
#include <hip/hip_bf16.h>

// ---------------------------------------------------------------------------
// Fused Broca module for MI455X (gfx1250).
// All heavy math as bf16 WMMA (v_wmma_f32_16x16x32_bf16, f32 accumulate).
// Weights are pre-swizzled ONCE into the exact per-lane B-fragment register
// image (bf16), so the GEMM inner loop is 1 WMMA : 1 aligned 32B load.
// Top-k gather is algebraically replaced by a gate-weighted dense sum
// (gate weights are zero for unselected experts), making every stage GEMM.
// ---------------------------------------------------------------------------

typedef __bf16 bf16;
typedef __attribute__((ext_vector_type(16))) __bf16 v16bf;
typedef __attribute__((ext_vector_type(8)))  __bf16 v8bf;
typedef __attribute__((ext_vector_type(8)))  float  v8f;

#define B_TOK 16384
#define D_DIM 256
#define S_DIM 64
#define H_DIM 512
#define E_NUM 8
#define DT_C  0.2f

// ---- WMMA wrapper: D = A(16x32 bf16) * B(32x16 bf16) + C(16x16 f32) -------
__device__ __forceinline__ v8f wmma_bf16(v16bf a, v16bf b, v8f c) {
  return __builtin_amdgcn_wmma_f32_16x16x32_bf16(
      /*neg_a=*/false, a, /*neg_b=*/false, b,
      /*c_mod=*/(short)0, c, /*reuse_a=*/false, /*reuse_b=*/false);
}

__device__ __forceinline__ v16bf join16(v8bf lo, v8bf hi) {
  return __builtin_shufflevector(lo, hi, 0, 1, 2, 3, 4, 5, 6, 7, 8, 9, 10, 11,
                                 12, 13, 14, 15);
}

// ---- A fragment (16x32) from row-major bf16 (LDS or global) ---------------
// ISA 7.12.2: lane<16: row=lane, K runs {k0..k0+7, k0+16..k0+23};
//             lane>=16: same rows, runs shifted by +8.  Two 16B loads.
__device__ __forceinline__ v16bf load_a(const bf16* p, long ld, long m0,
                                        int k0, int lane) {
  const int grp = lane >> 4;
  const long m  = m0 + (lane & 15);
  const bf16* base = p + m * ld + k0 + grp * 8;
  v8bf lo = *(const v8bf*)(base);
  v8bf hi = *(const v8bf*)(base + 16);
  return join16(lo, hi);
}

// ---- B fragment from pre-swizzled buffer: one aligned 32B load ------------
__device__ __forceinline__ v16bf load_b_frag(const bf16* fragbuf, long f,
                                             int lane) {
  return *(const v16bf*)(fragbuf + f * 512 + lane * 16);
}

// ===========================================================================
// Prep: swizzle f32 weight [batch][K][N] -> bf16 B-fragment layout.
// frag index f = (b*Kt + kt)*Nt + nt ; element (lane, e) holds
// W[b][kt*32 + 8*(lane>>4) + (e&7) + 16*(e>>3)][nt*16 + (lane&15)].
// ===========================================================================
__global__ __launch_bounds__(256) void k_prep(const float* __restrict__ W,
                                              bf16* __restrict__ out, int K,
                                              int N, long total) {
  long i = (long)blockIdx.x * 256 + threadIdx.x;
  if (i >= total) return;
  const int e    = i & 15;
  const int lane = (i >> 4) & 31;
  const long f   = i >> 9;
  const int Nt = N >> 4, Kt = K >> 5;
  const int nt = (int)(f % Nt);
  const long t2 = f / Nt;
  const int kt = (int)(t2 % Kt);
  const long b = t2 / Kt;
  const int k = kt * 32 + ((lane >> 4) << 3) + (e & 7) + ((e >> 3) << 4);
  const int n = nt * 16 + (lane & 15);
  out[i] = (bf16)W[(b * K + k) * (long)N + n];
}

// ===========================================================================
// Kernel 1: X = emb @ W_proj + b_proj (bf16), gate = top2-softmax(X@Wg+bg).
// 64 tokens / block, 4 wave32 waves, each wave owns 16 rows.
// ===========================================================================
__global__ __launch_bounds__(128) void k_proj_gate(
    const float* __restrict__ emb, const bf16* __restrict__ WpF,
    const float* __restrict__ bp, const float* __restrict__ Wg,
    const float* __restrict__ bg, bf16* __restrict__ Xout,
    float* __restrict__ Gout) {
  __shared__ bf16 sEmb[64 * 256];  // 32 KB
  __shared__ bf16 sX[64 * 64];     //  8 KB
  const int tid = threadIdx.x, lane = tid & 31, wave = tid >> 5;
  const long m0blk = (long)blockIdx.x * 64;

  for (int i = tid; i < 64 * 256; i += 128) {
    int r = i >> 8, c = i & 255;
    sEmb[i] = (bf16)emb[(m0blk + r) * D_DIM + c];
  }
  __syncthreads();

  const int mw = wave * 16, col = lane & 15, grp = lane >> 4;
  v8f zero = {};
  v8f acc[4] = {zero, zero, zero, zero};
#pragma unroll
  for (int kt = 0; kt < 8; ++kt) {          // K=256
    v16bf a = load_a(sEmb, 256, mw, kt * 32, lane);
#pragma unroll
    for (int nt = 0; nt < 4; ++nt)          // N=64
      acc[nt] = wmma_bf16(a, load_b_frag(WpF, kt * 4 + nt, lane), acc[nt]);
  }
#pragma unroll
  for (int nt = 0; nt < 4; ++nt) {
    const int n = nt * 16 + col;
    const float bv = bp[n];
#pragma unroll
    for (int r = 0; r < 8; ++r) {
      const int m = mw + r + grp * 8;
      bf16 h = (bf16)(acc[nt][r] + bv);
      sX[m * 64 + n] = h;
      Xout[(m0blk + m) * S_DIM + n] = h;
    }
  }
  __syncthreads();

  if (tid < 64) {  // one token per thread: logits, top-2, softmax
    float lg[E_NUM];
#pragma unroll
    for (int e = 0; e < E_NUM; ++e) {
      float s = bg[e];
      for (int k = 0; k < S_DIM; ++k)
        s += (float)sX[tid * 64 + k] * Wg[k * E_NUM + e];
      lg[e] = s;
    }
    int i0 = 0;
    for (int e = 1; e < E_NUM; ++e)
      if (lg[e] > lg[i0]) i0 = e;
    int i1 = (i0 == 0) ? 1 : 0;
    for (int e = 0; e < E_NUM; ++e)
      if (e != i0 && lg[e] > lg[i1]) i1 = e;
    const float e0 = __expf(lg[i0] - lg[i0]), e1 = __expf(lg[i1] - lg[i0]);
    const float inv = 1.f / (e0 + e1);
    float g[E_NUM];
#pragma unroll
    for (int e = 0; e < E_NUM; ++e) g[e] = 0.f;
    g[i0] = e0 * inv;
    g[i1] = e1 * inv;
#pragma unroll
    for (int e = 0; e < E_NUM; ++e) Gout[(m0blk + tid) * E_NUM + e] = g[e];
  }
}

// ===========================================================================
// Kernel 2: fused expert stage (2560 WMMA / wave).  Per 64-token block:
//   for e, for H-half:
//     pre = X @ W_in[e]           (A hoisted in regs, B = swizzled frags)
//     h   = g_e*DT*tanh(pre+b)/(1+DT/tau)  -> bf16, wave-private LDS rows
//     Cacc += h @ W_out[e]        (16 persistent f32 accum frags = 128 VGPR)
//   C = tanh(Cacc + g@b_out) -> f32 (d_out) + bf16 (scratch)
// pre/h never touch HBM (268 MB avoided).
// ===========================================================================
__global__ __launch_bounds__(128) void k_experts(
    const bf16* __restrict__ X, const float* __restrict__ G,
    const bf16* __restrict__ WinF, const float* __restrict__ bh,
    const float* __restrict__ tau, const bf16* __restrict__ WoutF,
    const float* __restrict__ bout, float* __restrict__ Cout,
    bf16* __restrict__ Cbf) {
  __shared__ bf16 sX[64 * 64];   //  8 KB
  __shared__ bf16 sH[64 * 256];  // 32 KB (one H-half; rows are wave-private)
  __shared__ float sG[64 * 8];   //  2 KB
  const int tid = threadIdx.x, lane = tid & 31, wave = tid >> 5;
  const long m0blk = (long)blockIdx.x * 64;

  for (int i = tid; i < 64 * 64; i += 128) sX[i] = X[m0blk * S_DIM + i];
  for (int i = tid; i < 64 * 8; i += 128) sG[i] = G[m0blk * E_NUM + i];
  __syncthreads();

  const int mw = wave * 16, col = lane & 15, grp = lane >> 4;
  v8f zero = {};
  v8f Cacc[16];
#pragma unroll
  for (int t = 0; t < 16; ++t) Cacc[t] = zero;

  // X-tile A fragments are invariant across all experts: hoist (16 VGPRs).
  v16bf aX0 = load_a(sX, 64, mw, 0, lane);
  v16bf aX1 = load_a(sX, 64, mw, 32, lane);

  for (int e = 0; e < E_NUM; ++e) {
    const bf16* WinE  = WinF + (long)e * S_DIM * H_DIM;   // Kt=2,  Nt=32
    const bf16* WoutE = WoutF + (long)e * H_DIM * D_DIM;  // Kt=16, Nt=16
    __builtin_prefetch(WinE, 0, 1);
    __builtin_prefetch(WoutE, 0, 1);
#pragma unroll
    for (int hh = 0; hh < 2; ++hh) {
      const int h0 = hh * 256;
      // --- pre GEMM + liquid epilogue into LDS ------------------------------
      for (int nt = 0; nt < 16; ++nt) {
        const int ng = hh * 16 + nt;  // global n-tile in [0,32)
        v8f p = wmma_bf16(aX0, load_b_frag(WinE, 0 * 32 + ng, lane), zero);
        p = wmma_bf16(aX1, load_b_frag(WinE, 1 * 32 + ng, lane), p);
        const int hcol = h0 + nt * 16 + col;  // tau/b_h are per-column
        const float bhv = bh[e * H_DIM + hcol];
        const float sc = DT_C / (1.f + DT_C / tau[e * H_DIM + hcol]);
#pragma unroll
        for (int r = 0; r < 8; ++r) {
          const int m = mw + r + grp * 8;
          const float ge = sG[m * 8 + e];  // sparse gate (0 if unselected)
          sH[m * 256 + hcol] = (bf16)(ge * sc * tanhf(p[r] + bhv));
        }
      }
      // --- out GEMM accumulate: Cacc += h @ W_out[e][h-half, :] ------------
      for (int kc = 0; kc < 8; ++kc) {
        v16bf a = load_a(sH, 256, mw, kc * 32, lane);
        const long kt = hh * 8 + kc;
#pragma unroll
        for (int nt = 0; nt < 16; ++nt)
          Cacc[nt] =
              wmma_bf16(a, load_b_frag(WoutE, kt * 16 + nt, lane), Cacc[nt]);
      }
    }
  }

  // epilogue: gate-weighted output bias, tanh, dual store
#pragma unroll
  for (int nt = 0; nt < 16; ++nt) {
    const int n = nt * 16 + col;
#pragma unroll
    for (int r = 0; r < 8; ++r) {
      const int m = mw + r + grp * 8;
      float bias = 0.f;
#pragma unroll
      for (int e = 0; e < E_NUM; ++e)
        bias += sG[m * 8 + e] * bout[e * D_DIM + n];
      const float c = tanhf(Cacc[nt][r] + bias);
      const long idx = (m0blk + m) * D_DIM + n;
      Cout[idx] = c;
      Cbf[idx] = (bf16)c;
    }
  }
}

// ===========================================================================
// Kernel 3: Titans memory MLP + surprise (kc-outer, 16 persistent accums).
// ===========================================================================
__global__ __launch_bounds__(128) void k_memory(
    const bf16* __restrict__ Cbf, const float* __restrict__ Cf,
    const bf16* __restrict__ W1F, const float* __restrict__ b1,
    const bf16* __restrict__ W2F, const float* __restrict__ b2,
    float* __restrict__ surprise) {
  __shared__ bf16 sT[64 * 256];  // 32 KB
  __shared__ float sS[64];
  const int tid = threadIdx.x, lane = tid & 31, wave = tid >> 5;
  const long m0blk = (long)blockIdx.x * 64;
  if (tid < 64) sS[tid] = 0.f;
  __syncthreads();

  const int mw = wave * 16, col = lane & 15, grp = lane >> 4;
  v8f zero = {};
  __builtin_prefetch(W1F, 0, 1);
  __builtin_prefetch(W2F, 0, 1);

  // t = tanh(C @ W1 + b1)      (A from global bf16: 2x16B loads per frag)
  v8f acc[16];
#pragma unroll
  for (int t = 0; t < 16; ++t) acc[t] = zero;
  for (int kc = 0; kc < 8; ++kc) {
    v16bf a = load_a(Cbf, D_DIM, m0blk + mw, kc * 32, lane);
#pragma unroll
    for (int nt = 0; nt < 16; ++nt)
      acc[nt] = wmma_bf16(a, load_b_frag(W1F, (long)kc * 16 + nt, lane),
                          acc[nt]);
  }
#pragma unroll
  for (int nt = 0; nt < 16; ++nt) {
    const int n = nt * 16 + col;
    const float bv = b1[n];
#pragma unroll
    for (int r = 0; r < 8; ++r)
      sT[(mw + r + grp * 8) * 256 + n] = (bf16)tanhf(acc[nt][r] + bv);
  }
  __syncthreads();

  // pred = t @ W2 + b2 ; squared error per row via ds_add_f32
#pragma unroll
  for (int t = 0; t < 16; ++t) acc[t] = zero;
  for (int kc = 0; kc < 8; ++kc) {
    v16bf a = load_a(sT, 256, mw, kc * 32, lane);
#pragma unroll
    for (int nt = 0; nt < 16; ++nt)
      acc[nt] = wmma_bf16(a, load_b_frag(W2F, (long)kc * 16 + nt, lane),
                          acc[nt]);
  }
  float part[8];
#pragma unroll
  for (int r = 0; r < 8; ++r) part[r] = 0.f;
#pragma unroll
  for (int nt = 0; nt < 16; ++nt) {
    const int n = nt * 16 + col;
    const float bv = b2[n];
#pragma unroll
    for (int r = 0; r < 8; ++r) {
      const int m = mw + r + grp * 8;
      const float d = (acc[nt][r] + bv) - Cf[(m0blk + m) * D_DIM + n];
      part[r] += d * d;
    }
  }
#pragma unroll
  for (int r = 0; r < 8; ++r) atomicAdd(&sS[mw + r + grp * 8], part[r]);
  __syncthreads();
  if (tid < 64) surprise[m0blk + tid] = sS[tid] * (1.f / D_DIM);
}

// ===========================================================================
extern "C" void kernel_launch(void* const* d_in, const int* in_sizes, int n_in,
                              void* d_out, int out_size, void* d_ws,
                              size_t ws_size, hipStream_t stream) {
  (void)in_sizes; (void)n_in; (void)out_size; (void)ws_size;
  const float* emb  = (const float*)d_in[0];   // [B, D]
  const float* Wp   = (const float*)d_in[1];   // [D, S]
  const float* bp   = (const float*)d_in[2];   // [S]
  const float* Wg   = (const float*)d_in[3];   // [S, E]
  const float* bg   = (const float*)d_in[4];   // [E]
  const float* Win  = (const float*)d_in[5];   // [E, S, H]
  const float* bh   = (const float*)d_in[6];   // [E, H]
  const float* tau  = (const float*)d_in[7];   // [E, H]
  const float* Wout = (const float*)d_in[8];   // [E, H, D]
  const float* bout = (const float*)d_in[9];   // [E, D]
  const float* W1   = (const float*)d_in[10];  // [D, D]
  const float* b1   = (const float*)d_in[11];  // [D]
  const float* W2   = (const float*)d_in[12];  // [D, D]
  const float* b2   = (const float*)d_in[13];  // [D]

  // ---- workspace layout (~14 MB) ----
  char* ws = (char*)d_ws;
  size_t off = 0;
  bf16* Xbuf = (bf16*)(ws + off); off += (size_t)B_TOK * S_DIM * 2;   // 2 MB
  float* Gbuf = (float*)(ws + off); off += (size_t)B_TOK * E_NUM * 4; // .5 MB
  bf16* Cbuf = (bf16*)(ws + off); off += (size_t)B_TOK * D_DIM * 2;   // 8 MB
  bf16* WpF  = (bf16*)(ws + off); off += (size_t)D_DIM * S_DIM * 2;
  bf16* WinF = (bf16*)(ws + off); off += (size_t)E_NUM * S_DIM * H_DIM * 2;
  bf16* WoutF = (bf16*)(ws + off); off += (size_t)E_NUM * H_DIM * D_DIM * 2;
  bf16* W1F  = (bf16*)(ws + off); off += (size_t)D_DIM * D_DIM * 2;
  bf16* W2F  = (bf16*)(ws + off); off += (size_t)D_DIM * D_DIM * 2;

  float* combined = (float*)d_out;                     // [B, D] f32
  float* surprise = combined + (size_t)B_TOK * D_DIM;  // [B] f32

  // ---- weight swizzle (once per launch; ~1.4M elems total) ----
  auto blocks = [](long n) { return (unsigned)((n + 255) / 256); };
  long tWp = (long)D_DIM * S_DIM;
  long tWi = (long)E_NUM * S_DIM * H_DIM;
  long tWo = (long)E_NUM * H_DIM * D_DIM;
  long tW1 = (long)D_DIM * D_DIM;
  k_prep<<<blocks(tWp), 256, 0, stream>>>(Wp, WpF, D_DIM, S_DIM, tWp);
  k_prep<<<blocks(tWi), 256, 0, stream>>>(Win, WinF, S_DIM, H_DIM, tWi);
  k_prep<<<blocks(tWo), 256, 0, stream>>>(Wout, WoutF, H_DIM, D_DIM, tWo);
  k_prep<<<blocks(tW1), 256, 0, stream>>>(W1, W1F, D_DIM, D_DIM, tW1);
  k_prep<<<blocks(tW1), 256, 0, stream>>>(W2, W2F, D_DIM, D_DIM, tW1);

  dim3 blk(128);
  dim3 grd(B_TOK / 64);
  k_proj_gate<<<grd, blk, 0, stream>>>(emb, WpF, bp, Wg, bg, Xbuf, Gbuf);
  k_experts<<<grd, blk, 0, stream>>>(Xbuf, Gbuf, WinF, bh, tau, WoutF, bout,
                                     combined, Cbuf);
  k_memory<<<grd, blk, 0, stream>>>(Cbuf, combined, W1F, b1, W2F, b2,
                                    surprise);
}
